// RecursiveNN_82162724372980
// MI455X (gfx1250) — matrix operand order repaired and checked by reading, not compiled, not run on hardware
//
#include <hip/hip_runtime.h>

typedef __attribute__((ext_vector_type(16))) _Float16 v16h;
typedef __attribute__((ext_vector_type(8)))  _Float16 v8h;
typedef __attribute__((ext_vector_type(4)))  _Float16 v4h;
typedef __attribute__((ext_vector_type(8)))  float    v8f;
typedef __attribute__((ext_vector_type(4)))  float    f32x4;

#define WSTR 104   // half (2B) stride of channel rows: 208 B = 52 dwords, 16B-aligned, bank-friendly

__device__ __forceinline__ int wrapi(int v, int n) {
  return v < 0 ? v + n : (v >= n ? v - n : v);
}

// A fragment (16x32, M x K), 16-bit: lane = row (lane&15); laneLo holds K 0..7 & 16..23,
// laneHi holds K 8..15 & 24..31 (per CDNA5 ISA 16-bit A layout).
__device__ __forceinline__ v16h load_fragA(const _Float16* row, int k, int laneHi) {
  const int lo = 32 * k + (laneHi ? 8 : 0);
  v16h f;
  ((f32x4*)&f)[0] = *(const f32x4*)(row + lo);
  ((f32x4*)&f)[1] = *(const f32x4*)(row + lo + 16);
  return f;
}

// B fragment (32x16, K x N), 16-bit: lane&15 = column N; laneLo holds K 0..7 & 16..23,
// laneHi holds K 8..15 & 24..31 (same order as A).
__device__ __forceinline__ v16h load_fragB(const _Float16* row, int k, int laneHi) {
  const int lo = 32 * k + (laneHi ? 8 : 0);
  v16h f;
  ((f32x4*)&f)[0] = *(const f32x4*)(row + lo);
  ((f32x4*)&f)[1] = *(const f32x4*)(row + lo + 16);
  return f;
}

// fixed perception filters (cross-correlation, matching lax.conv)
__device__ __forceinline__ void stencil4(float a00,float a01,float a02,
                                         float a10,float a11,float a12,
                                         float a20,float a21,float a22,
                                         float& id,float& sx,float& sy,float& lp) {
  id = a11;
  sx = (a02 - a00) + 2.f*(a12 - a10) + (a22 - a20);
  sy = (a20 - a00) + 2.f*(a21 - a01) + (a22 - a02);
  lp = (a00 + a02 + a20 + a22) + 2.f*(a01 + a10 + a12 + a21) - 12.f*a11;
}

// One fused resolution level:
//   perceive(x) [+ bilinear-up(perceive(xc))] -> 1x1 conv (bias, relu) -> 1x1 conv [+ residual]
// H: fine resolution; CIN1: input channels of first 1x1 conv; K1PAD: CIN1 padded to x32.
template <int H, int CIN1, int K1PAD, bool HAS_UP>
__global__ __launch_bounds__(256)
void nca_level_kernel(const float* __restrict__ x,      // [B,12,H,H]
                      const float* __restrict__ xc,     // [B,12,H/2,H/2] or null
                      const float* __restrict__ w1,     // [96,CIN1]
                      const float* __restrict__ b1,     // [96]
                      const float* __restrict__ w2,     // [12,96]
                      float* __restrict__ out,          // [B,12,H,H]
                      const float* __restrict__ resid)  // residual (x) or null
{
  constexpr int C   = 12;
  constexpr int Hc  = H / 2;
  constexpr int KC1 = K1PAD / 32;
  const int tid = threadIdx.x;
  const int b   = blockIdx.z;
  const int Y0 = blockIdx.y * 16, X0 = blockIdx.x * 16;
  const int cy0 = blockIdx.y * 8, cx0 = blockIdx.x * 8;

  extern __shared__ __align__(16) char smem[];
  float* xs    = (float*)smem;                         // [12][18][20] fine tile + halo
  float* x1s   = xs + 12*18*20;                        // [12][12][14] coarse tile + halo
  float* z1s   = x1s + (HAS_UP ? 12*12*14 : 0);        // [100][48] coarse perceive
  float* biasS = z1s + (HAS_UP ? 100*48 : 0);          // [96]
  _Float16* g   = (_Float16*)(biasS + 96);             // [256][WSTR] activations (pixel-major)
  _Float16* Wl1 = g + 256*WSTR;                        // [96][WSTR]
  _Float16* Wl2 = Wl1 + 96*WSTR;                       // [16][WSTR]

  // ---- phase 1: stage inputs (circular halo) and f16 weights into LDS ----
  for (int i = tid; i < 12*18*18; i += 256) {
    int c = i / 324; int rem = i - c*324; int r = rem / 18; int cl = rem - r*18;
    int gy = wrapi(Y0 - 1 + r, H), gx = wrapi(X0 - 1 + cl, H);
    xs[(c*18 + r)*20 + cl] = x[((size_t)(b*C + c)*H + gy)*H + gx];
  }
  if constexpr (HAS_UP) {
    for (int i = tid; i < 12*12*12; i += 256) {
      int c = i / 144; int rem = i - c*144; int r = rem / 12; int cl = rem - r*12;
      int gy = wrapi(cy0 - 2 + r, Hc), gx = wrapi(cx0 - 2 + cl, Hc);
      x1s[(c*12 + r)*14 + cl] = xc[((size_t)(b*C + c)*Hc + gy)*Hc + gx];
    }
  }
  for (int i = tid; i < 96*K1PAD; i += 256) {
    int o = i / K1PAD; int ci = i - o*K1PAD;
    Wl1[o*WSTR + ci] = (ci < CIN1) ? (_Float16)w1[o*CIN1 + ci] : (_Float16)0.f;
  }
  for (int i = tid; i < 16*96; i += 256) {
    int o = i / 96; int ci = i - o*96;
    Wl2[o*WSTR + ci] = (o < 12) ? (_Float16)w2[o*96 + ci] : (_Float16)0.f;
  }
  if (tid < 96) biasS[tid] = b1[tid];
  __syncthreads();

  // ---- phase 2: perceive fine -> g[px][0..47]; coarse perceive + bilinear up -> g[px][48..95] ----
  const int px = tid, ty = px >> 4, tx = px & 15;
  #pragma unroll
  for (int c = 0; c < 12; ++c) {
    const float* a = &xs[(c*18 + ty)*20 + tx];
    float id, sx, sy, lp;
    stencil4(a[0],a[1],a[2], a[20],a[21],a[22], a[40],a[41],a[42], id,sx,sy,lp);
    v4h q; q[0]=(_Float16)id; q[1]=(_Float16)sx; q[2]=(_Float16)sy; q[3]=(_Float16)lp;
    *(v4h*)&g[px*WSTR + 4*c] = q;
  }
  if constexpr (!HAS_UP) {   // pad K to 64 with zeros for the 96x48 GEMM
    v8h zz;
    #pragma unroll
    for (int j = 0; j < 8; ++j) zz[j] = (_Float16)0.f;
    *(v8h*)&g[px*WSTR + 48] = zz;
    *(v8h*)&g[px*WSTR + 56] = zz;
  }
  if constexpr (HAS_UP) {
    for (int i = tid; i < 100*12; i += 256) {          // z_coarse on 10x10 region (+1 halo)
      int pos = i / 12; int c = i - pos*12;
      int r = pos / 10; int cl = pos - r*10;
      const float* a = &x1s[(c*12 + r)*14 + cl];
      float id, sx, sy, lp;
      stencil4(a[0],a[1],a[2], a[14],a[15],a[16], a[28],a[29],a[30], id,sx,sy,lp);
      f32x4 q = {id, sx, sy, lp};
      *(f32x4*)&z1s[pos*48 + 4*c] = q;
    }
    __syncthreads();
    // jax.image.resize bilinear, 2x: out coord = i/2 - 0.25; edge renormalize == clamp here
    const int Y = Y0 + ty, X = X0 + tx;
    const int jy = Y >> 1, jx = X >> 1;
    const int t0y = (Y & 1) ? jy : jy - 1;
    const int t0x = (X & 1) ? jx : jx - 1;
    const float w0y = (Y & 1) ? 0.75f : 0.25f;
    const float w0x = (X & 1) ? 0.75f : 0.25f;
    const float w1y = 1.f - w0y, w1x = 1.f - w0x;
    const int l0y = min(max(t0y,     0), Hc-1) - cy0 + 1;
    const int l1y = min(max(t0y + 1, 0), Hc-1) - cy0 + 1;
    const int l0x = min(max(t0x,     0), Hc-1) - cx0 + 1;
    const int l1x = min(max(t0x + 1, 0), Hc-1) - cx0 + 1;
    const float* z00 = &z1s[(l0y*10 + l0x)*48];
    const float* z01 = &z1s[(l0y*10 + l1x)*48];
    const float* z10 = &z1s[(l1y*10 + l0x)*48];
    const float* z11 = &z1s[(l1y*10 + l1x)*48];
    const float c00 = w0y*w0x, c01 = w0y*w1x, c10 = w1y*w0x, c11 = w1y*w1x;
    #pragma unroll
    for (int c = 0; c < 12; ++c) {
      f32x4 v = c00 * *(const f32x4*)(z00 + 4*c)
              + c01 * *(const f32x4*)(z01 + 4*c)
              + c10 * *(const f32x4*)(z10 + 4*c)
              + c11 * *(const f32x4*)(z11 + 4*c);
      v4h q;
      #pragma unroll
      for (int j = 0; j < 4; ++j) q[j] = (_Float16)v[j];
      *(v4h*)&g[px*WSTR + 48 + 4*c] = q;
    }
  }
  __syncthreads();

  // ---- phase 3: per-wave WMMA GEMM chain ----
  const int lane = tid & 31;
  const int laneHi = lane >> 4;
  const int ln = lane & 15;
  const int wv = tid >> 5;
  for (int nt = wv; nt < 16; nt += 8) {
    const int pixBase = nt * 16;
    const _Float16* grow = &g[(pixBase + ln) * WSTR];
    // B fragments for GEMM1 (held in regs so in-place relu overwrite below is safe)
    v16h fb0 = load_fragB(grow, 0, laneHi);
    v16h fb1 = load_fragB(grow, 1, laneHi);
    v16h fb2 = fb0;
    if constexpr (KC1 == 3) fb2 = load_fragB(grow, 2, laneHi);
    #pragma unroll
    for (int mt = 0; mt < 6; ++mt) {
      v8f acc;
      #pragma unroll
      for (int v = 0; v < 8; ++v) acc[v] = biasS[16*mt + laneHi*8 + v];
      const _Float16* wrow = &Wl1[(16*mt + ln) * WSTR];
      acc = __builtin_amdgcn_wmma_f32_16x16x32_f16(false, load_fragA(wrow,0,laneHi),
                                                   false, fb0, (short)0, acc, false, false);
      acc = __builtin_amdgcn_wmma_f32_16x16x32_f16(false, load_fragA(wrow,1,laneHi),
                                                   false, fb1, (short)0, acc, false, false);
      if constexpr (KC1 == 3)
        acc = __builtin_amdgcn_wmma_f32_16x16x32_f16(false, load_fragA(wrow,2,laneHi),
                                                     false, fb2, (short)0, acc, false, false);
      // relu, convert to f16, write back in-place (C/D layout: lane&15 = pixel, VGPR v = out-ch)
      v8h t;
      #pragma unroll
      for (int v = 0; v < 8; ++v) t[v] = (_Float16)fmaxf(acc[v], 0.f);
      *(v8h*)&g[(pixBase + ln)*WSTR + 16*mt + laneHi*8] = t;
    }
    // GEMM2: 12x96 (rows 12..15 zero) over the relu'd activations
    v8f o;
    #pragma unroll
    for (int v = 0; v < 8; ++v) o[v] = 0.f;
    const _Float16* w2row = &Wl2[ln * WSTR];
    o = __builtin_amdgcn_wmma_f32_16x16x32_f16(false, load_fragA(w2row,0,laneHi),
                                               false, load_fragB(grow,0,laneHi), (short)0, o, false, false);
    o = __builtin_amdgcn_wmma_f32_16x16x32_f16(false, load_fragA(w2row,1,laneHi),
                                               false, load_fragB(grow,1,laneHi), (short)0, o, false, false);
    o = __builtin_amdgcn_wmma_f32_16x16x32_f16(false, load_fragA(w2row,2,laneHi),
                                               false, load_fragB(grow,2,laneHi), (short)0, o, false, false);
    const int p  = pixBase + ln;
    const int oy = Y0 + (p >> 4), ox = X0 + (p & 15);
    #pragma unroll
    for (int v = 0; v < 8; ++v) {
      int ch = laneHi*8 + v;
      if (ch < 12) {
        size_t idx = ((size_t)(b*C + ch)*H + oy)*H + ox;
        float val = o[v];
        if constexpr (!HAS_UP) val += resid[idx];
        out[idx] = val;
      }
    }
  }
}

extern "C" void kernel_launch(void* const* d_in, const int* in_sizes, int n_in,
                              void* d_out, int out_size, void* d_ws, size_t ws_size,
                              hipStream_t stream) {
  (void)in_sizes; (void)n_in; (void)out_size; (void)d_ws; (void)ws_size;
  const float* x0   = (const float*)d_in[0];
  const float* x1   = (const float*)d_in[1];
  const float* x2   = (const float*)d_in[2];
  const float* w0_1 = (const float*)d_in[3];
  const float* b0_1 = (const float*)d_in[4];
  const float* w0_2 = (const float*)d_in[5];
  const float* w1_1 = (const float*)d_in[6];
  const float* b1_1 = (const float*)d_in[7];
  const float* w1_2 = (const float*)d_in[8];
  const float* w2_1 = (const float*)d_in[9];
  const float* b2_1 = (const float*)d_in[10];
  const float* w2_2 = (const float*)d_in[11];

  float* out0 = (float*)d_out;
  float* out1 = out0 + (size_t)16*12*256*256;
  float* out2 = out1 + (size_t)16*12*128*128;

  const size_t smemUp = (size_t)(12*18*20 + 12*12*14 + 100*48 + 96)*4
                      + (size_t)(256 + 96 + 16)*WSTR*2;   // ~121.5 KB
  const size_t smemNo = (size_t)(12*18*20 + 96)*4
                      + (size_t)(256 + 96 + 16)*WSTR*2;   // ~94.2 KB

  nca_level_kernel<256, 96, 96, true ><<<dim3(16,16,16), 256, smemUp, stream>>>(
      x0, x1, w0_1, b0_1, w0_2, out0, nullptr);
  nca_level_kernel<128, 96, 96, true ><<<dim3(8,8,16),   256, smemUp, stream>>>(
      x1, x2, w1_1, b1_1, w1_2, out1, nullptr);
  nca_level_kernel< 64, 48, 64, false><<<dim3(4,4,16),   256, smemNo, stream>>>(
      x2, nullptr, w2_1, b2_1, w2_2, out2, x2);
}
